// SelfAttention_50878182588822
// MI455X (gfx1250) — compile-verified, matrix operand order checked by reading, not run on hardware
//
#include <hip/hip_runtime.h>

// Problem constants (from reference): B=4, C=256, H=W=64 -> N=4096, C8=32
#define BB    4
#define CC    256
#define NN    4096
#define CQK   32

typedef __attribute__((ext_vector_type(16))) __bf16    v16bf;
typedef __attribute__((ext_vector_type(8)))  __bf16    v8bf;
typedef __attribute__((ext_vector_type(8)))  float     v8f;
typedef __attribute__((ext_vector_type(8)))  unsigned  v8u;

// Round-and-truncate f32 -> bf16 (round-half-up; 2 VALU)
static __device__ __forceinline__ __bf16 f2bf(float f) {
  unsigned u = __builtin_bit_cast(unsigned, f) + 0x8000u;
  unsigned short s = (unsigned short)(u >> 16);
  return __builtin_bit_cast(__bf16, s);
}

// Pack two f32 into one dword of bf16 (a -> low16, b -> high16) via v_perm_b32.
static __device__ __forceinline__ unsigned pack_bf16x2(float a, float b) {
  unsigned ua = __builtin_bit_cast(unsigned, a) + 0x8000u;
  unsigned ub = __builtin_bit_cast(unsigned, b) + 0x8000u;
  // dst.b0=ua.b2(sel 6), dst.b1=ua.b3(sel 7), dst.b2=ub.b2(sel 2), dst.b3=ub.b3(sel 3)
  return __builtin_amdgcn_perm(ua, ub, 0x03020706u);
}

static __device__ __forceinline__ v16bf cat8(v8bf lo, v8bf hi) {
  return __builtin_shufflevector(lo, hi, 0,1,2,3,4,5,6,7,8,9,10,11,12,13,14,15);
}

static __device__ __forceinline__ v8f wmma_bf16(v16bf a, v16bf b, v8f c) {
  return __builtin_amdgcn_wmma_f32_16x16x32_bf16(false, a, false, b, (short)0, c,
                                                 false, false);
}

// ---------------------------------------------------------------------------
// Kernel 0: x [B][C][N] f32  ->  xT [B][N][C] bf16   (LDS tiled transpose)
// ---------------------------------------------------------------------------
__global__ __launch_bounds__(256) void k_transpose_cvt(
    const float* __restrict__ x, __bf16* __restrict__ xT) {
  __shared__ float tile[16][17];
  const int n0 = blockIdx.x * 16, c0 = blockIdx.y * 16, b = blockIdx.z;
  const int tx = threadIdx.x, ty = threadIdx.y;
  tile[ty][tx] = x[((size_t)b * CC + (c0 + ty)) * NN + n0 + tx];
  __syncthreads();
  xT[((size_t)b * NN + (n0 + ty)) * CC + c0 + tx] = f2bf(tile[tx][ty]);
}

// ---------------------------------------------------------------------------
// Kernel 1: projections via WMMA.
//   fT[B][N][32], gT[B][N][32] (pre-transposed for attention fragments)
//   hv[B][256][N]              (row-contiguous in N for PV A fragments)
// grid = (N/16, 5, B), block = 128 (4 waves). o-tile = blockIdx.y*4 + wave:
//   0,1 -> Wq rows 0..31 ; 2,3 -> Wk rows 0..31 ; 4..19 -> Wv rows 0..255
// ---------------------------------------------------------------------------
__global__ __launch_bounds__(128) void k_proj(
    const __bf16* __restrict__ xT,
    const float*  __restrict__ Wq,
    const float*  __restrict__ Wk,
    const float*  __restrict__ Wv,
    __bf16* __restrict__ fT,
    __bf16* __restrict__ gT,
    __bf16* __restrict__ hv) {
  const int lane = threadIdx.x & 31;
  const int wave = threadIdx.x >> 5;
  const int hl   = lane >> 4;          // lane-half
  const int ln   = lane & 15;
  const int koff = hl * 8;
  const int n0   = blockIdx.x * 16;
  const int b    = blockIdx.z;
  const int ot   = blockIdx.y * 4 + wave;      // 0..19

  const float* W;
  int orow;
  if (ot < 2)      { W = Wq; orow = ot * 16; }
  else if (ot < 4) { W = Wk; orow = (ot - 2) * 16; }
  else             { W = Wv; orow = (ot - 4) * 16; }

  const float*  wp = W  + (size_t)(orow + ln) * CC;             // A row (f32)
  const __bf16* xp = xT + ((size_t)b * NN + n0 + ln) * CC;      // B row (bf16)

  v8f acc = {};
#pragma unroll
  for (int k0 = 0; k0 < CC; k0 += 32) {
    v8u au;
#pragma unroll
    for (int d = 0; d < 4; ++d)
      au[d] = pack_bf16x2(wp[k0 + koff + 2 * d], wp[k0 + koff + 2 * d + 1]);
#pragma unroll
    for (int d = 0; d < 4; ++d)
      au[d + 4] =
          pack_bf16x2(wp[k0 + koff + 16 + 2 * d], wp[k0 + koff + 17 + 2 * d]);
    v16bf A = __builtin_bit_cast(v16bf, au);
    v8bf xlo = *(const v8bf*)(xp + k0 + koff);
    v8bf xhi = *(const v8bf*)(xp + k0 + koff + 16);
    acc = wmma_bf16(A, cat8(xlo, xhi), acc);
  }

  if (ot < 4) {
    // D lane l, reg r: o = orow + r + 8*hl, n = n0 + ln.  Pack 8 bf16 -> b128.
    __bf16* dst = (ot < 2) ? fT : gT;
    const int obase = 16 * (ot & 1) + 8 * hl;
    uint4 o4;
    o4.x = pack_bf16x2(acc[0], acc[1]);
    o4.y = pack_bf16x2(acc[2], acc[3]);
    o4.z = pack_bf16x2(acc[4], acc[5]);
    o4.w = pack_bf16x2(acc[6], acc[7]);
    *(uint4*)(dst + ((size_t)b * NN + n0 + ln) * CQK + obase) = o4;
  } else {
    const int cb = (ot - 4) * 16 + 8 * hl;
#pragma unroll
    for (int r = 0; r < 8; ++r)
      hv[((size_t)b * CC + cb + r) * NN + n0 + ln] = f2bf(acc[r]);
  }
}

// ---------------------------------------------------------------------------
// Kernel 2: flash attention.
// grid = (N/64, B), block = 512 (16 waves). wave = 4*cg + jt:
//   jt in 0..3 selects the 16-column j-tile, cg in 0..3 owns channels
//   [64*cg, 64*cg+64). Each wave redundantly computes S + online softmax for
//   its j-tile (no inter-wave sync; 64 columns/block quarters L2 V-traffic).
// S tile:  A = fT rows (16 i x 32 c), B = gT rows-as-columns (32 c x 16 j).
// PV tile: A = hv   (16 c x 32 i),    B = P (32 i x 16 j) built lane-locally
//          from the two S-tile f32 accumulators (D->B layout coincidence).
// ---------------------------------------------------------------------------
__global__ __launch_bounds__(512) void k_attn(
    const __bf16* __restrict__ fT,
    const __bf16* __restrict__ gT,
    const __bf16* __restrict__ hv,
    const float*  __restrict__ xres,
    const float*  __restrict__ gamma,
    float* __restrict__ out) {
  const int lane  = threadIdx.x & 31;
  const int wave  = threadIdx.x >> 5;
  const int jt    = wave & 3;
  const int cg    = wave >> 2;
  const int hl    = lane >> 4;
  const int ln    = lane & 15;
  const int koff  = hl * 8;
  const int j0    = blockIdx.x * 64 + jt * 16;
  const int b     = blockIdx.y;
  const int cbase = cg * 64;

  // Loop-invariant B operand: g^T column block (32 c x 16 j)
  const __bf16* gp = gT + ((size_t)b * NN + j0 + ln) * CQK;
  const v16bf Bg = cat8(*(const v8bf*)(gp + koff), *(const v8bf*)(gp + koff + 16));

  const __bf16* fTb = fT + (size_t)b * NN * CQK;
  const __bf16* hvb = hv + (size_t)b * CC * (size_t)NN;

  v8f acc[4] = {v8f{}, v8f{}, v8f{}, v8f{}};
  float m_run = -__builtin_inff();
  float s_run = 0.f;

  for (int i0 = 0; i0 < NN; i0 += 32) {
    if (i0 + 32 < NN) {   // warm L2/L0 for next chunk (global_prefetch_b8)
      __builtin_prefetch(fTb + (size_t)(i0 + 32 + ln) * CQK, 0, 3);
      __builtin_prefetch(hvb + (size_t)(cbase + ln) * NN + i0 + 32, 0, 3);
    }

    // Keys: two A fragments of f^T (rows i0..i0+15 and i0+16..i0+31)
    const __bf16* fp0 = fTb + (size_t)(i0 + ln) * CQK;
    const __bf16* fp1 = fTb + (size_t)(i0 + 16 + ln) * CQK;
    v16bf A0 = cat8(*(const v8bf*)(fp0 + koff), *(const v8bf*)(fp0 + koff + 16));
    v16bf A1 = cat8(*(const v8bf*)(fp1 + koff), *(const v8bf*)(fp1 + koff + 16));
    v8f S0 = wmma_bf16(A0, Bg, v8f{});
    v8f S1 = wmma_bf16(A1, Bg, v8f{});

    // Online softmax over i for this lane's column j = j0 + ln.
    float mt = -__builtin_inff();
#pragma unroll
    for (int e = 0; e < 8; ++e) mt = fmaxf(mt, fmaxf(S0[e], S1[e]));
    mt = fmaxf(mt, __shfl_xor(mt, 16, 32));   // partner lane has other i-half
    const float mn = fmaxf(m_run, mt);

    float corr = 1.0f;
    if (mt > m_run) {           // max grew for this column: rescale (rare)
      corr = __expf(m_run - mn);
#pragma unroll
      for (int ct = 0; ct < 4; ++ct)
#pragma unroll
        for (int e = 0; e < 8; ++e) acc[ct][e] *= corr;
    }
    m_run = mn;

    float ssum = 0.f;
    v8u pu;                     // B operand of PV wmma, built lane-locally
#pragma unroll
    for (int d = 0; d < 4; ++d) {
      float a0 = __expf(S0[2 * d] - mn);
      float a1 = __expf(S0[2 * d + 1] - mn);
      float b0 = __expf(S1[2 * d] - mn);
      float b1 = __expf(S1[2 * d + 1] - mn);
      ssum += (a0 + a1) + (b0 + b1);
      pu[d]     = pack_bf16x2(a0, a1);
      pu[d + 4] = pack_bf16x2(b0, b1);
    }
    const v16bf P = __builtin_bit_cast(v16bf, pu);
    ssum += __shfl_xor(ssum, 16, 32);
    s_run = s_run * corr + ssum;

    // Values: 4 channel tiles of 16, K = 32 i's
#pragma unroll
    for (int ct = 0; ct < 4; ++ct) {
      const __bf16* vp = hvb + (size_t)(cbase + ct * 16 + ln) * NN + i0 + koff;
      v16bf Av = cat8(*(const v8bf*)vp, *(const v8bf*)(vp + 16));
      acc[ct] = wmma_bf16(Av, P, acc[ct]);
    }
  }

  const float inv = 1.0f / s_run;
  const float gm  = gamma[0];
#pragma unroll
  for (int ct = 0; ct < 4; ++ct)
#pragma unroll
    for (int r = 0; r < 8; ++r) {
      const int c = cbase + ct * 16 + 8 * hl + r;
      const size_t idx = ((size_t)b * CC + c) * NN + j0 + ln;
      out[idx] = gm * (acc[ct][r] * inv) + xres[idx];
    }
}

// ---------------------------------------------------------------------------
extern "C" void kernel_launch(void* const* d_in, const int* in_sizes, int n_in,
                              void* d_out, int out_size, void* d_ws, size_t ws_size,
                              hipStream_t stream) {
  (void)in_sizes; (void)n_in; (void)out_size; (void)ws_size;
  const float* x     = (const float*)d_in[0];
  const float* Wq    = (const float*)d_in[1];
  const float* Wk    = (const float*)d_in[2];
  const float* Wv    = (const float*)d_in[3];
  const float* gamma = (const float*)d_in[4];
  float*       out   = (float*)d_out;

  char* ws = (char*)d_ws;
  const size_t XT_BYTES = (size_t)BB * NN * CC * 2;   // 8 MB
  const size_t FT_BYTES = (size_t)BB * NN * CQK * 2;  // 1 MB
  __bf16* xT = (__bf16*)(ws);
  __bf16* fT = (__bf16*)(ws + XT_BYTES);
  __bf16* gT = (__bf16*)(ws + XT_BYTES + FT_BYTES);
  __bf16* hv = (__bf16*)(ws + XT_BYTES + 2 * FT_BYTES);

  k_transpose_cvt<<<dim3(NN / 16, CC / 16, BB), dim3(16, 16), 0, stream>>>(x, xT);
  k_proj<<<dim3(NN / 16, 5, BB), 128, 0, stream>>>(xT, Wq, Wk, Wv, fT, gT, hv);
  k_attn<<<dim3(NN / 64, BB), 512, 0, stream>>>(fT, gT, hv, x, gamma, out);
}
